// MambaEncoder_28527172780218
// MI455X (gfx1250) — matrix operand order repaired and checked by reading, not compiled, not run on hardware
//
#include <hip/hip_runtime.h>
#include <hip/hip_bf16.h>
#include <math.h>

// Problem constants (from reference)
#define NLAYER 4
#define DMODEL 1024
#define DINNER 2048
#define DSTATE 16
#define DCONV  4
#define DRANK  64
#define BATCH  2
#define SEQ    2048
#define ROWS   (BATCH*SEQ)        // 4096
#define XZCOLS (2*DINNER)         // 4096
#define XDBLCOLS (DRANK+2*DSTATE) // 96

// GEMM tiling
#define BM 128
#define BN 64
#define BK 32
#define SAW 18   // dwords per LDS row: 16 data (32 bf16) + 2 pad (conflict-free, 8B-aligned)

typedef __attribute__((ext_vector_type(16))) __bf16 bf16x16;
typedef __attribute__((ext_vector_type(8)))  float  floatx8;

// fp32 pair -> packed bf16 (round-to-nearest, ties up: 5 cheap VALU ops)
static __device__ __forceinline__ unsigned int pack2bf(float a, float b) {
  unsigned int ua = __float_as_uint(a) + 0x8000u;
  unsigned int ub = __float_as_uint(b) + 0x8000u;
  return (ua >> 16) | (ub & 0xFFFF0000u);
}

union Frag { unsigned int u[8]; bf16x16 v; };

// Phase 1: issue global float4 loads for one K-step (no LDS writes -> no waits)
static __device__ __forceinline__ void load_tiles(
    const float* __restrict__ A, const float* __restrict__ W,
    int lda, int K, int N, int rowBase, int colBase, int k0, int tid,
    float4* ra, float4* rw)
{
  #pragma unroll
  for (int i = 0; i < 4; ++i) {
    int f = tid + i * 256;          // float4 index
    int r = f >> 3, c4 = f & 7;     // 8 float4 per 32-float row
    ra[i] = *(const float4*)&A[(size_t)(rowBase + r) * lda + k0 + c4 * 4];
  }
  #pragma unroll
  for (int i = 0; i < 2; ++i) {
    int f = tid + i * 256;
    int r = f >> 3, c4 = f & 7;
    int wr = colBase + r; if (wr >= N) wr = N - 1;   // clamp; stores guarded later
    rw[i] = *(const float4*)&W[(size_t)wr * (size_t)K + k0 + c4 * 4];
  }
}

// Phase 2: pack registers to bf16 and store into LDS tile buffers
static __device__ __forceinline__ void store_tiles(
    const float4* ra, const float4* rw, int tid,
    unsigned int* sAbuf, unsigned int* sWbuf)
{
  #pragma unroll
  for (int i = 0; i < 4; ++i) {
    int f = tid + i * 256;
    int r = f >> 3, c4 = f & 7;
    sAbuf[r * SAW + c4 * 2]     = pack2bf(ra[i].x, ra[i].y);
    sAbuf[r * SAW + c4 * 2 + 1] = pack2bf(ra[i].z, ra[i].w);
  }
  #pragma unroll
  for (int i = 0; i < 2; ++i) {
    int f = tid + i * 256;
    int r = f >> 3, c4 = f & 7;
    sWbuf[r * SAW + c4 * 2]     = pack2bf(rw[i].x, rw[i].y);
    sWbuf[r * SAW + c4 * 2 + 1] = pack2bf(rw[i].z, rw[i].w);
  }
}

// ---------------------------------------------------------------------------
// C[M,N] = A[M,K] (fp32, row stride lda) x W[N,K]^T (fp32 row-major), bf16 WMMA
// mode 0: plain store; mode 1: softplus(C + bias[n])  (for dt_proj)
// Block: 256 threads = 8 waves (4x2). Block tile 128x64; each wave computes a
// 2x2 grid of 16x16 WMMA tiles -> 4 v_wmma per K-step.
// Register-buffered + LDS double-buffered pipeline:
//   issue ldg(k+1) -> compute WMMA(k) -> pack/ds_store(k+1) -> barrier
// so the global-load waits land AFTER the matrix math. L2 prefetch of k+2.
// ---------------------------------------------------------------------------
__global__ __launch_bounds__(256) void gemm_bf16_wmma(
    const float* __restrict__ A, const float* __restrict__ W,
    float* __restrict__ C, const float* __restrict__ bias,
    int M, int N, int K, int lda, int ldc, int mode)
{
  __shared__ unsigned int sA[2][BM * SAW];   // [buf][m][k-pair] bf16x2
  __shared__ unsigned int sW[2][BN * SAW];   // [buf][n][k-pair] bf16x2

  const int tid  = threadIdx.x;
  const int lane = tid & 31;
  const int wave = tid >> 5;
  const int hi   = lane >> 4;         // upper half-wave -> k 8-15 and 24-31
  const int l15  = lane & 15;
  const int m0   = (wave & 3) * 32;   // wave's two M-tiles: m0, m0+16
  const int n0   = (wave >> 2) * 32;  // wave's two N-tiles: n0, n0+16
  const int rowBase = blockIdx.y * BM;
  const int colBase = blockIdx.x * BN;

  floatx8 acc[2][2];
  #pragma unroll
  for (int mi = 0; mi < 2; ++mi)
    #pragma unroll
    for (int ni = 0; ni < 2; ++ni)
      acc[mi][ni] = (floatx8){0.f,0.f,0.f,0.f,0.f,0.f,0.f,0.f};

  float4 ra[4], rw[2];

  // Prologue: fill buffer 0
  load_tiles(A, W, lda, K, N, rowBase, colBase, 0, tid, ra, rw);
  store_tiles(ra, rw, tid, sA[0], sW[0]);
  __syncthreads();

  int cur = 0;
  for (int k0 = 0; k0 < K; k0 += BK) {
    const int nxt = cur ^ 1;
    const bool have_next = (k0 + BK < K);

    // Issue next tile's global loads (consumed only after the WMMAs)
    if (have_next)
      load_tiles(A, W, lda, K, N, rowBase, colBase, k0 + BK, tid, ra, rw);
    // Prefetch tile k+2 toward L2/WGP$ (global_prefetch_b8)
    if (k0 + 2 * BK < K) {
      __builtin_prefetch(&A[(size_t)(rowBase + (tid >> 3)) * lda + (k0 + 2 * BK) + (tid & 7) * 4], 0, 3);
      int rwc = colBase + (tid >> 3); if (rwc >= N) rwc = N - 1;
      __builtin_prefetch(&W[(size_t)rwc * (size_t)K + (k0 + 2 * BK) + (tid & 7) * 4], 0, 3);
    }

    // A fragments: lane row m; dword j holds K pair; idx = j + (j>=4?4:0) + hi*4
    Frag fa[2], fb[2];
    #pragma unroll
    for (int mi = 0; mi < 2; ++mi) {
      const unsigned int* pa = &sA[cur][(m0 + mi * 16 + l15) * SAW];
      #pragma unroll
      for (int j = 0; j < 8; ++j)
        fa[mi].u[j] = pa[j + ((j >= 4) ? 4 : 0) + hi * 4];
    }
    // B fragments: lane col n; dword j -> K pair idx = j + (j>=4?4:0) + hi*4
    #pragma unroll
    for (int ni = 0; ni < 2; ++ni) {
      const unsigned int* pw = &sW[cur][(n0 + ni * 16 + l15) * SAW];
      #pragma unroll
      for (int j = 0; j < 8; ++j)
        fb[ni].u[j] = pw[j + ((j >= 4) ? 4 : 0) + hi * 4];
    }

    #pragma unroll
    for (int mi = 0; mi < 2; ++mi)
      #pragma unroll
      for (int ni = 0; ni < 2; ++ni)
        acc[mi][ni] = __builtin_amdgcn_wmma_f32_16x16x32_bf16(
            false, fa[mi].v, false, fb[ni].v,
            (short)0, acc[mi][ni], false, false);

    // Pack+store next tile to the alternate LDS buffer (load waits land here)
    if (have_next)
      store_tiles(ra, rw, tid, sA[nxt], sW[nxt]);

    __syncthreads();   // next-tile stores visible AND this-tile reads done
    cur = nxt;
  }

  // C/D layout: lane l, dword j -> (M = j + hi*8, N = l&15)
  #pragma unroll
  for (int ni = 0; ni < 2; ++ni) {
    const int gn = colBase + n0 + ni * 16 + l15;
    if (gn >= N) continue;
    float bv = (mode == 1) ? bias[gn] : 0.f;
    #pragma unroll
    for (int mi = 0; mi < 2; ++mi) {
      #pragma unroll
      for (int j = 0; j < 8; ++j) {
        int gm = rowBase + m0 + mi * 16 + j + hi * 8;
        float v = acc[mi][ni][j];
        if (mode == 1) {
          v += bv;
          // softplus via HW transcendentals (v_exp_f32 / v_log_f32)
          v = (v > 20.f) ? v : __logf(1.f + __expf(v));
        }
        C[(size_t)gm * ldc + gn] = v;
      }
    }
  }
}

// ---------------------------------------------------------------------------
// Residual accumulate + LayerNorm over D=1024. One block (256 thr) per row.
// res = h_in (+ res);  out = LN(res) * w + b
// ---------------------------------------------------------------------------
__global__ __launch_bounds__(256) void ln_residual(
    const float* __restrict__ h_in, float* __restrict__ res,
    const float* __restrict__ w, const float* __restrict__ b,
    float* __restrict__ out, int first)
{
  const int row = blockIdx.x;
  const int tid = threadIdx.x;
  float v[4]; float s = 0.f, s2 = 0.f;
  #pragma unroll
  for (int k = 0; k < 4; ++k) {
    int c = tid + k * 256;
    float x = h_in[(size_t)row * DMODEL + c];
    if (!first) x += res[(size_t)row * DMODEL + c];
    res[(size_t)row * DMODEL + c] = x;
    v[k] = x; s += x; s2 += x * x;
  }
  #pragma unroll
  for (int m = 16; m >= 1; m >>= 1) {
    s  += __shfl_xor(s,  m, 32);
    s2 += __shfl_xor(s2, m, 32);
  }
  __shared__ float rs[8], rs2[8];
  if ((tid & 31) == 0) { rs[tid >> 5] = s; rs2[tid >> 5] = s2; }
  __syncthreads();
  if (tid == 0) {
    float t = 0.f, t2 = 0.f;
    #pragma unroll
    for (int j = 0; j < 8; ++j) { t += rs[j]; t2 += rs2[j]; }
    rs[0] = t; rs2[0] = t2;
  }
  __syncthreads();
  const float mean = rs[0] * (1.f / DMODEL);
  const float var  = rs2[0] * (1.f / DMODEL) - mean * mean;
  const float inv  = rsqrtf(var + 1e-5f);
  #pragma unroll
  for (int k = 0; k < 4; ++k) {
    int c = tid + k * 256;
    out[(size_t)row * DMODEL + c] = (v[k] - mean) * inv * w[c] + b[c];
  }
}

// ---------------------------------------------------------------------------
// Depthwise causal conv (DC=4) + bias + SiLU; reads xi = xz[:, 0:DINNER]
// ---------------------------------------------------------------------------
__global__ void conv_silu(const float* __restrict__ xz, const float* __restrict__ cw,
                          const float* __restrict__ cb, float* __restrict__ xc)
{
  int idx = blockIdx.x * blockDim.x + threadIdx.x;   // over ROWS*DINNER
  if (idx >= ROWS * DINNER) return;
  int d   = idx % DINNER;
  int row = idx / DINNER;       // row = b*SEQ + l
  int l   = row % SEQ;
  float acc = cb[d];
  #pragma unroll
  for (int j = 0; j < DCONV; ++j) {
    int t = l - (DCONV - 1) + j;
    if (t >= 0)
      acc += xz[(size_t)(row - (DCONV - 1) + j) * XZCOLS + d] * cw[d * DCONV + j];
  }
  float sg = 1.f / (1.f + __expf(-acc));
  xc[(size_t)row * DINNER + d] = acc * sg;           // SiLU
}

// ---------------------------------------------------------------------------
// Selective scan: one state per lane (16 lanes per channel, 2 channels/wave).
// h_s(t) = exp(dt*A_s)*h_s + dt*B_s*x;  y = sum_s h_s*C_s + x*D
// Cross-state reduction via 4 xor-shuffle steps within the 16-lane group.
// ---------------------------------------------------------------------------
__global__ __launch_bounds__(256) void selective_scan(
    const float* __restrict__ xc, const float* __restrict__ dt,
    const float* __restrict__ xdbl, const float* __restrict__ A_log,
    const float* __restrict__ Dp, float* __restrict__ y)
{
  int gid = blockIdx.x * 256 + threadIdx.x;  // BATCH*DINNER*DSTATE = 65536
  int s   = gid & (DSTATE - 1);
  int ch  = gid >> 4;
  int d   = ch & (DINNER - 1);
  int b   = ch >> 11;
  const float A  = -__expf(A_log[d * DSTATE + s]);
  const float Dv = Dp[d];
  float h = 0.f;
  for (int t = 0; t < SEQ; ++t) {
    size_t row = (size_t)(b * SEQ + t);
    float dtv = dt[row * DINNER + d];
    float xv  = xc[row * DINNER + d];
    float Bv  = xdbl[row * XDBLCOLS + DRANK + s];
    float Cv  = xdbl[row * XDBLCOLS + DRANK + DSTATE + s];
    float dA  = __expf(dtv * A);
    h = fmaf(dA, h, dtv * Bv * xv);
    float p = h * Cv;
    #pragma unroll
    for (int m = 8; m >= 1; m >>= 1) p += __shfl_xor(p, m, 32);  // stays in 16-group
    if (s == 0) y[row * DINNER + d] = p + xv * Dv;
  }
}

// y *= SiLU(z), z = xz[:, DINNER + d]
__global__ void gate_silu(float* __restrict__ y, const float* __restrict__ xz)
{
  int idx = blockIdx.x * blockDim.x + threadIdx.x;
  if (idx >= ROWS * DINNER) return;
  int d   = idx % DINNER;
  int row = idx / DINNER;
  float z  = xz[(size_t)row * XZCOLS + DINNER + d];
  float sg = 1.f / (1.f + __expf(-z));
  y[idx] *= z * sg;
}

// ---------------------------------------------------------------------------
extern "C" void kernel_launch(void* const* d_in, const int* in_sizes, int n_in,
                              void* d_out, int out_size, void* d_ws, size_t ws_size,
                              hipStream_t stream)
{
  (void)in_sizes; (void)n_in; (void)out_size; (void)ws_size;
  const float* hidden     = (const float*)d_in[0];
  const float* in_proj_w  = (const float*)d_in[1];
  const float* conv_w     = (const float*)d_in[2];
  const float* conv_b     = (const float*)d_in[3];
  const float* x_proj_w   = (const float*)d_in[4];
  const float* dt_proj_w  = (const float*)d_in[5];
  const float* dt_proj_b  = (const float*)d_in[6];
  const float* A_log      = (const float*)d_in[7];
  const float* D_param    = (const float*)d_in[8];
  const float* out_proj_w = (const float*)d_in[9];
  const float* ln_w       = (const float*)d_in[10];
  const float* ln_b       = (const float*)d_in[11];
  const float* normf_w    = (const float*)d_in[12];
  const float* normf_b    = (const float*)d_in[13];

  // Workspace layout (fp32): ~220 MB
  float* p = (float*)d_ws;
  float* buf_res  = p; p += (size_t)ROWS * DMODEL;
  float* buf_hn   = p; p += (size_t)ROWS * DMODEL;
  float* buf_h    = p; p += (size_t)ROWS * DMODEL;
  float* buf_xz   = p; p += (size_t)ROWS * XZCOLS;
  float* buf_xc   = p; p += (size_t)ROWS * DINNER;
  float* buf_xdbl = p; p += (size_t)ROWS * XDBLCOLS;
  float* buf_dt   = p; p += (size_t)ROWS * DINNER;
  float* buf_y    = p; p += (size_t)ROWS * DINNER;

  dim3 blk(256);
  const int ne = ROWS * DINNER;

  for (int i = 0; i < NLAYER; ++i) {
    const float* hsrc = (i == 0) ? hidden : buf_h;
    // residual += h; hn = LN(residual)
    ln_residual<<<ROWS, blk, 0, stream>>>(hsrc, buf_res,
        ln_w + (size_t)i * DMODEL, ln_b + (size_t)i * DMODEL, buf_hn, i == 0 ? 1 : 0);
    // xz = hn @ in_proj_w^T   [4096 x 4096]
    gemm_bf16_wmma<<<dim3(XZCOLS/BN, ROWS/BM), blk, 0, stream>>>(
        buf_hn, in_proj_w + (size_t)i * XZCOLS * DMODEL, buf_xz, nullptr,
        ROWS, XZCOLS, DMODEL, DMODEL, XZCOLS, 0);
    // xc = SiLU(depthwise_causal_conv(xi) + conv_b)
    conv_silu<<<(ne + 255)/256, blk, 0, stream>>>(buf_xz,
        conv_w + (size_t)i * DINNER * DCONV, conv_b + (size_t)i * DINNER, buf_xc);
    // xdbl = xc @ x_proj_w^T   [4096 x 96]
    gemm_bf16_wmma<<<dim3((XDBLCOLS + BN - 1)/BN, ROWS/BM), blk, 0, stream>>>(
        buf_xc, x_proj_w + (size_t)i * XDBLCOLS * DINNER, buf_xdbl, nullptr,
        ROWS, XDBLCOLS, DINNER, DINNER, XDBLCOLS, 0);
    // dt = softplus(xdbl[:, :64] @ dt_proj_w^T + dt_proj_b)   [4096 x 2048]
    gemm_bf16_wmma<<<dim3(DINNER/BN, ROWS/BM), blk, 0, stream>>>(
        buf_xdbl, dt_proj_w + (size_t)i * DINNER * DRANK, buf_dt,
        dt_proj_b + (size_t)i * DINNER,
        ROWS, DINNER, DRANK, XDBLCOLS, DINNER, 1);
    // y = selective_scan(xc, dt, B, C, A, D)
    selective_scan<<<(BATCH*DINNER*DSTATE)/256, blk, 0, stream>>>(
        buf_xc, buf_dt, buf_xdbl,
        A_log + (size_t)i * DINNER * DSTATE, D_param + (size_t)i * DINNER, buf_y);
    // y *= SiLU(z)
    gate_silu<<<(ne + 255)/256, blk, 0, stream>>>(buf_y, buf_xz);
    // h = y @ out_proj_w^T   [4096 x 1024]
    gemm_bf16_wmma<<<dim3(DMODEL/BN, ROWS/BM), blk, 0, stream>>>(
        buf_y, out_proj_w + (size_t)i * DMODEL * DINNER, buf_h, nullptr,
        ROWS, DMODEL, DINNER, DINNER, DMODEL, 0);
  }
  // final: residual += h; out = LN(residual) with normf
  ln_residual<<<ROWS, blk, 0, stream>>>(buf_h, buf_res, normf_w, normf_b,
                                        (float*)d_out, 0);
}
